// OriginalSpectralResidualNorm_68401649156411
// MI455X (gfx1250) — matrix operand order repaired and checked by reading, not compiled, not run on hardware
//
#include <hip/hip_runtime.h>

// CDNA5 / gfx1250: wave32, WMMA 16x16x32 bf16 -> f32
typedef __attribute__((ext_vector_type(16))) __bf16 v16bf;
typedef __attribute__((ext_vector_type(8)))  float  v8f;
typedef __attribute__((ext_vector_type(4)))  int    v4i_;

#define GLOBAL_AS __attribute__((address_space(1)))
#define LDS_AS    __attribute__((address_space(3)))

#define HDIM 384
#define WDIM 384
#define NCH  6
#define IMG  (HDIM * WDIM)
#define TROWS 16        // output rows per workgroup
#define SLAB_ROWS 64    // rows [r0-16, r0+47] (two K=32 slabs, zero-padded)
#define SPITCH 392      // slab row pitch in bf16 (>=384, mult of 8)
#define CPITCH 448      // colsum pitch in bf16: 16 left zero pad + 384 + 48 right zero pad
#define INV441 (1.0f / 441.0f)

// Async global<->LDS path (CDNA5 ASYNCcnt engine) — probe via __has_builtin,
// fall back to plain VGPR copy if this toolchain doesn't declare the builtins.
#if __has_builtin(__builtin_amdgcn_global_load_async_to_lds_b128) && \
    __has_builtin(__builtin_amdgcn_global_store_async_from_lds_b128) && \
    __has_builtin(__builtin_amdgcn_s_wait_asynccnt)
#define USE_ASYNC_COPY 1
#else
#define USE_ASYNC_COPY 0
#endif

// One workgroup computes residual for a 16-row x 384-col stripe of one mag channel.
// Stage 0: cooperative fill of bf16 log1p slab (64 rows incl. halo, zero outside image).
// Pass 1 (WMMA): column box-sum, banded-ones A (const, in regs) x data B (LDS gather).
// Pass 2 (WMMA): row box-sum, data A (LDS, 16B vector loads) x banded-ones B (const).
// Output: residual = log1p(mag) [f32, re-read, L2-hot] - rowsum/441.
__global__ __launch_bounds__(256) void spectral_residual_kernel(
    const float* __restrict__ in, float* __restrict__ out)
{
    __shared__ __bf16 slab[SLAB_ROWS * SPITCH];      // 50,176 B
    __shared__ __bf16 colsum[TROWS * CPITCH];        // 14,336 B

    const int tid  = threadIdx.x;
    const int lane = tid & 31;
    const int wave = tid >> 5;     // 0..7
    const int n    = lane & 15;    // column-within-tile / A-row index
    const int hi   = lane >> 4;    // lane-half select

    const int wg  = blockIdx.x;                 // 32*3*24 blocks
    const int b   = wg / (3 * 24);
    const int rem = wg % (3 * 24);
    const int ch  = rem / 24;                   // mag channel 0..2
    const int rt  = rem % 24;
    const int r0  = rt * TROWS;
    const size_t imgBase = (size_t)(b * NCH + ch) * IMG;
    const float* mag = in + imgBase;
    float*       res = out + imgBase;

    // ---------- stage 0: fill slab with bf16(log1p(mag)), zeros outside image ----------
    #pragma unroll
    for (int it = 0; it < (SLAB_ROWS * (WDIM / 4)) / 256; ++it) {  // 24 iters
        int chunk = tid + it * 256;
        int s  = chunk / (WDIM / 4);
        int c4 = (chunk % (WDIM / 4)) * 4;
        int g  = r0 - 16 + s;
        __bf16* dst = &slab[s * SPITCH + c4];
        if ((unsigned)g < (unsigned)HDIM) {
            const float4 v = *(const float4*)(mag + (size_t)g * WDIM + c4);
            dst[0] = (__bf16)log1pf(v.x);
            dst[1] = (__bf16)log1pf(v.y);
            dst[2] = (__bf16)log1pf(v.z);
            dst[3] = (__bf16)log1pf(v.w);
        } else {
            dst[0] = (__bf16)0.0f; dst[1] = (__bf16)0.0f;
            dst[2] = (__bf16)0.0f; dst[3] = (__bf16)0.0f;
        }
    }
    // zero colsum pads: phys cols [0,16) and [400,448) for all 16 rows
    for (int q = tid; q < TROWS * 64; q += 256) {
        int row = q >> 6;
        int pc  = q & 63;
        int col = (pc < 16) ? pc : (pc + 384);
        colsum[row * CPITCH + col] = (__bf16)0.0f;
    }
    __syncthreads();

    // ---------- constant banded-ones A for column pass ----------
    // A 16x32, this code's k order (same as B below): lane m (0..15)/(16..31), element e -> K = 16*hi + e
    v16bf a0, a1;
    {
        const int m = n;
        #pragma unroll
        for (int e = 0; e < 16; ++e) {
            int k = (hi << 4) + e;
            // out row m reads slab rows K in [m+6, m+26]; slab1 covers K+32
            a0[e] = (k >= m + 6 && k <= m + 26) ? (__bf16)1.0f : (__bf16)0.0f;
            a1[e] = (k <= m - 6)                ? (__bf16)1.0f : (__bf16)0.0f;
        }
    }

    // ---------- pass 1: column box-sums via WMMA, store bf16 into colsum ----------
    #pragma unroll
    for (int tt = 0; tt < 3; ++tt) {
        const int j  = wave + tt * 8;   // N-tile 0..23
        const int c0 = j * 16;
        v8f acc = {0.f, 0.f, 0.f, 0.f, 0.f, 0.f, 0.f, 0.f};
        v16bf bm;
        // B 32x16, this code's k order (same as A): lane = n + 16*kHi, element e -> K = 16*kHi + e
        #pragma unroll
        for (int e = 0; e < 16; ++e)
            bm[e] = slab[(16 * hi + e) * SPITCH + c0 + n];
        acc = __builtin_amdgcn_wmma_f32_16x16x32_bf16(false, a0, false, bm,
                                                      (short)0, acc, false, false);
        #pragma unroll
        for (int e = 0; e < 16; ++e)
            bm[e] = slab[(32 + 16 * hi + e) * SPITCH + c0 + n];
        acc = __builtin_amdgcn_wmma_f32_16x16x32_bf16(false, a1, false, bm,
                                                      (short)0, acc, false, false);
        // C/D layout: VGPR v, lanes 0-15 -> M=v, lanes 16-31 -> M=v+8; col = n
        #pragma unroll
        for (int v = 0; v < 8; ++v)
            colsum[(v + 8 * hi) * CPITCH + 16 + c0 + n] = (__bf16)acc[v];
    }
    __syncthreads();

    // ---------- constant banded-ones B for row pass ----------
    v16bf b0, b1;
    #pragma unroll
    for (int e = 0; e < 16; ++e) {
        int k = e + (e & 8) + (hi << 3);
        b0[e] = (k >= n + 6 && k <= n + 26) ? (__bf16)1.0f : (__bf16)0.0f;
        int K = 32 + k;
        b1[e] = (K >= n + 6 && K <= n + 26) ? (__bf16)1.0f : (__bf16)0.0f;
    }

    // ---------- pass 2: row box-sums via WMMA, emit residual ----------
    #pragma unroll
    for (int tt = 0; tt < 3; ++tt) {
        const int j = wave + tt * 8;
        v8f acc = {0.f, 0.f, 0.f, 0.f, 0.f, 0.f, 0.f, 0.f};
        // A gather: row m = n, phys cols 16j + K; elements 0-7 <-> K=8*hi+0..7,
        // elements 8-15 <-> K=16+8*hi+0..7  => two aligned 16-byte LDS loads per slab
        const __bf16* arow = &colsum[n * CPITCH + 16 * j + 8 * hi];
        union { uint4 u; __bf16 h[8]; } p0, p1;
        v16bf am;
        p0.u = *(const uint4*)(arow);
        p1.u = *(const uint4*)(arow + 16);
        #pragma unroll
        for (int e = 0; e < 8; ++e) { am[e] = p0.h[e]; am[8 + e] = p1.h[e]; }
        acc = __builtin_amdgcn_wmma_f32_16x16x32_bf16(false, am, false, b0,
                                                      (short)0, acc, false, false);
        p0.u = *(const uint4*)(arow + 32);
        p1.u = *(const uint4*)(arow + 48);
        #pragma unroll
        for (int e = 0; e < 8; ++e) { am[e] = p0.h[e]; am[8 + e] = p1.h[e]; }
        acc = __builtin_amdgcn_wmma_f32_16x16x32_bf16(false, am, false, b1,
                                                      (short)0, acc, false, false);
        // residual = log1p(mag) - env ; re-read mag (resident in 192MB L2), f32 log1p
        #pragma unroll
        for (int v = 0; v < 8; ++v) {
            int row = v + 8 * hi;                 // 0..15
            int col = 16 * j + n;                 // 0..383
            size_t idx = (size_t)(r0 + row) * WDIM + col;
            float lm = log1pf(mag[idx]);
            res[idx] = lm - acc[v] * INV441;
        }
    }
}

// Phase channels (3..5) are a straight 56.6 MB each-way stream.
#if USE_ASYNC_COPY
// CDNA5 path: bounce 128b/lane through LDS with the async engine (ASYNCcnt).
// Deep outstanding queue, no VGPR writeback dependency; latency hidden across
// the 8 waves per block. 4 KB in flight per wave-burst, 16 KB LDS stage.
__global__ __launch_bounds__(256) void phase_copy_kernel(
    const float4* __restrict__ in4, float4* __restrict__ out4, int total4)
{
    __shared__ alignas(16) char stage[256 * 4 * 16];   // 16 KB, 4 slots per thread
    const int tid = threadIdx.x;
    const int PER_BATCH4 = 3 * (IMG / 4);      // 110592 float4 per batch
    const int CH3_OFF4   = IMG / 4;            // 36864 float4 per channel
    for (int base = blockIdx.x * 1024; base < total4; base += gridDim.x * 1024) {
        #pragma unroll
        for (int p = 0; p < 4; ++p) {
            int idx = base + p * 256 + tid;
            if (idx < total4) {
                int bb = idx / PER_BATCH4;
                int r  = idx % PER_BATCH4;
                size_t off = (size_t)(bb * NCH + 3) * CH3_OFF4 + r;
                __builtin_amdgcn_global_load_async_to_lds_b128(
                    (GLOBAL_AS v4i_*)(in4 + off),
                    (LDS_AS v4i_*)(stage + (p * 256 + tid) * 16),
                    0, 0);
            }
        }
        __builtin_amdgcn_s_wait_asynccnt(0);   // loads landed in LDS
        #pragma unroll
        for (int p = 0; p < 4; ++p) {
            int idx = base + p * 256 + tid;
            if (idx < total4) {
                int bb = idx / PER_BATCH4;
                int r  = idx % PER_BATCH4;
                size_t off = (size_t)(bb * NCH + 3) * CH3_OFF4 + r;
                __builtin_amdgcn_global_store_async_from_lds_b128(
                    (GLOBAL_AS v4i_*)(out4 + off),
                    (LDS_AS v4i_*)(stage + (p * 256 + tid) * 16),
                    0, 0);
            }
        }
        __builtin_amdgcn_s_wait_asynccnt(0);   // stage safe to overwrite next iter
    }
}
#else
// Fallback: plain float4 grid-stride copy.
__global__ __launch_bounds__(256) void phase_copy_kernel(
    const float4* __restrict__ in4, float4* __restrict__ out4, int total4)
{
    const int PER_BATCH4 = 3 * (IMG / 4);      // 110592 float4 per batch
    const int CH3_OFF4   = IMG / 4;            // 36864 float4 per channel
    for (int idx = blockIdx.x * 256 + threadIdx.x; idx < total4;
         idx += gridDim.x * 256) {
        int bb = idx / PER_BATCH4;
        int r  = idx % PER_BATCH4;
        size_t off = (size_t)(bb * NCH + 3) * CH3_OFF4 + r;
        out4[off] = in4[off];
    }
}
#endif

extern "C" void kernel_launch(void* const* d_in, const int* in_sizes, int n_in,
                              void* d_out, int out_size, void* d_ws, size_t ws_size,
                              hipStream_t stream) {
    (void)in_sizes; (void)n_in; (void)d_ws; (void)ws_size; (void)out_size;
    const float* in = (const float*)d_in[0];   // fft_coeffs (32,6,384,384) f32
    float* out = (float*)d_out;                // (32,6,384,384) f32

    // mag residual: 32 batches * 3 channels * 24 row-stripes
    spectral_residual_kernel<<<32 * 3 * 24, 256, 0, stream>>>(in, out);

    // phase passthrough
    const int total4 = 32 * 3 * (IMG / 4);
    phase_copy_kernel<<<2048, 256, 0, stream>>>((const float4*)in, (float4*)out, total4);
}
